// ChebyshevLayer_80719615361565
// MI455X (gfx1250) — matrix operand order repaired and checked by reading, not compiled, not run on hardware
//
#include <hip/hip_runtime.h>

typedef _Float16 v16h __attribute__((ext_vector_type(16)));
typedef _Float16 v4h  __attribute__((ext_vector_type(4)));
typedef float    v8f  __attribute__((ext_vector_type(8)));
typedef unsigned u32x4 __attribute__((ext_vector_type(4)));
typedef float    f32x4 __attribute__((ext_vector_type(4)));
typedef int      v4i_gcc __attribute__((__vector_size__(16)));  // matches builtin proto

#define BDIM   8192
#define INDIM  1024
#define OUTDIM 1024
#define NSEG   11          // seg 0 = residual (raw x @ W), segs 1..10 = T_1..T_10
#define BM     128
#define BN     64
#define BK     32
#define NCHUNK (INDIM / BK)
#define APAD   36          // padded row length (halves) for t/x tiles
#define NTHREADS 128       // 4 waves; each wave owns 32m x 64n (Mw=2, Nw=4)

#if defined(__has_builtin)
#if __has_builtin(__builtin_amdgcn_global_load_async_to_lds_b128) && \
    __has_builtin(__builtin_amdgcn_s_wait_asynccnt)
#define HAVE_ASYNC_LDS 1
#endif
#endif

// ---------------------------------------------------------------------------
// fast tanh: prefer gfx1250 V_TANH hardware op
// ---------------------------------------------------------------------------
__device__ __forceinline__ _Float16 fast_tanh_f16(float xv) {
#if defined(__has_builtin) && __has_builtin(__builtin_amdgcn_tanh_f16)
  return __builtin_amdgcn_tanh_f16((_Float16)xv);
#elif defined(__has_builtin) && __has_builtin(__builtin_amdgcn_tanhf)
  return (_Float16)__builtin_amdgcn_tanhf(xv);
#else
  return (_Float16)tanhf(xv);
#endif
}

// ---------------------------------------------------------------------------
// Kernel 1: repack weights into K-major f16:  wBt[s][o][i]
// ---------------------------------------------------------------------------
__global__ __launch_bounds__(256) void cheb_transform_weights(
    const float* __restrict__ coeffs, const float* __restrict__ basew,
    _Float16* __restrict__ wBt) {
  __shared__ _Float16 tile[64][66];
  const int i0 = blockIdx.x * 64;
  const int o0 = blockIdx.y * 64;
  for (int s = 0; s < NSEG; ++s) {
    #pragma unroll
    for (int e = 0; e < 16; ++e) {
      int idx = threadIdx.x + e * 256;
      int il = idx >> 6, ol = idx & 63;
      float v = (s == 0)
          ? basew[(size_t)(i0 + il) * OUTDIM + (o0 + ol)]
          : coeffs[((size_t)(i0 + il) * OUTDIM + (o0 + ol)) * NSEG + s];
      tile[il][ol] = (_Float16)v;
    }
    __syncthreads();
    #pragma unroll
    for (int e = 0; e < 16; ++e) {
      int idx = threadIdx.x + e * 256;
      int ol = idx >> 6, il = idx & 63;
      wBt[(size_t)s * OUTDIM * INDIM + (size_t)(o0 + ol) * INDIM + (i0 + il)] =
          tile[il][ol];
    }
    __syncthreads();
  }
}

// ---------------------------------------------------------------------------
// Kernel 2: bias[o] = sum_i coeffs[i][o][0]   (the T_0 == 1 GEMM folded away)
// ---------------------------------------------------------------------------
__global__ __launch_bounds__(256) void cheb_bias(
    const float* __restrict__ coeffs, float* __restrict__ bias) {
  const int o = blockIdx.x;
  float s = 0.f;
  for (int i = threadIdx.x; i < INDIM; i += 256)
    s += coeffs[((size_t)i * OUTDIM + o) * NSEG];
  __shared__ float red[256];
  red[threadIdx.x] = s;
  __syncthreads();
  for (int st = 128; st > 0; st >>= 1) {
    if (threadIdx.x < st) red[threadIdx.x] += red[threadIdx.x + st];
    __syncthreads();
  }
  if (threadIdx.x == 0) bias[o] = red[0];
}

// ---------------------------------------------------------------------------
// Kernel 3: fused Chebyshev GEMM, double-buffered async B staging.
// ---------------------------------------------------------------------------
__device__ __forceinline__ v16h splat16(_Float16 s) {
  v16h r;
  #pragma unroll
  for (int i = 0; i < 16; ++i) r[i] = s;
  return r;
}

__global__ __launch_bounds__(NTHREADS) void cheb_gemm(
    const float* __restrict__ x, const _Float16* __restrict__ wBt,
    const float* __restrict__ bias, float* __restrict__ out) {
  __shared__ _Float16 sTh[BM][APAD];               // tanh(x) tile, f16
  __shared__ _Float16 sXh[BM][APAD];               // raw x tile, f16
  __shared__ _Float16 sB[2][NSEG][BN][BK];         // ping-pong weight tiles

  const int m0 = blockIdx.y * BM;
  const int n0 = blockIdx.x * BN;

  const int lane  = threadIdx.x & 31;
  const int w     = threadIdx.x >> 5;              // 0..3
  const int mbase = w * 32;
  const int lm    = lane & 15;
  const int khalf = lane >> 4;                     // 0 or 1
  const int koffA = khalf * 8;                     // A-fragment K offset

  // per-thread B staging coordinates (11 segs x 2 b128 chunks each)
  const int ci0 = threadIdx.x;                     // pass 0 chunk id
  const int ci1 = threadIdx.x + NTHREADS;          // pass 1 chunk id

  auto stage_B = [&](int buf, int ckSrc) {
    const int i0s = ckSrc * BK;
    #pragma unroll
    for (int p = 0; p < 2; ++p) {
      const int ci = p ? ci1 : ci0;
      const int n  = ci >> 2;
      const int k0 = (ci & 3) * 8;
      const _Float16* src = wBt + (size_t)(n0 + n) * INDIM + i0s + k0;
      #pragma unroll
      for (int s = 0; s < NSEG; ++s) {
#ifdef HAVE_ASYNC_LDS
        __builtin_amdgcn_global_load_async_to_lds_b128(
            (v4i_gcc*)(src + (size_t)s * OUTDIM * INDIM),
            (v4i_gcc*)&sB[buf][s][n][k0], 0, 0);
#else
        *(u32x4*)&sB[buf][s][n][k0] =
            *(const u32x4*)(src + (size_t)s * OUTDIM * INDIM);
#endif
      }
    }
  };

  v8f acc[2][4];
  #pragma unroll
  for (int mt = 0; mt < 2; ++mt)
    #pragma unroll
    for (int nt = 0; nt < 4; ++nt)
      acc[mt][nt] = v8f{0.f, 0.f, 0.f, 0.f, 0.f, 0.f, 0.f, 0.f};

  // prologue: kick off chunk 0's weight tiles
  stage_B(0, 0);

  for (int ck = 0; ck < NCHUNK; ++ck) {
    const int i0 = ck * BK;
    __syncthreads();                 // compute(ck-1) done -> LDS reusable

    // issue next chunk's async weight loads into the alternate buffer
    stage_B((ck + 1) & 1, (ck + 1) & (NCHUNK - 1));

    // ---- stage x tile: load f32, hw tanh, packed f16 b64 stores -----------
    {
      #pragma unroll
      for (int p = 0; p < 8; ++p) {
        const int f4 = threadIdx.x + p * NTHREADS;   // 1024 float4 groups
        const int r  = f4 >> 3;
        const int c0 = (f4 & 7) * 4;
        f32x4 v = *(const f32x4*)(x + (size_t)(m0 + r) * INDIM + i0 + c0);
        v4h xh, th;
        #pragma unroll
        for (int e = 0; e < 4; ++e) {
          xh[e] = (_Float16)v[e];
          th[e] = fast_tanh_f16(v[e]);
        }
        *(v4h*)&sXh[r][c0] = xh;
        *(v4h*)&sTh[r][c0] = th;
      }
    }
#ifdef HAVE_ASYNC_LDS
    __builtin_amdgcn_s_wait_asynccnt(22);  // only ck+1's 22 loads may remain
#endif
    __syncthreads();

    const _Float16 (*bb)[BN][BK] = sB[ck & 1];
    auto ldB = [&](int s, int nt) -> v16h {
      union { v16h v; u32x4 q[2]; } f;
      const _Float16* p = &bb[s][nt * 16 + lm][khalf * 8];
      f.q[0] = *(const u32x4*)p;
      f.q[1] = *(const u32x4*)(p + 16);
      return f.v;
    };

    // ---- gather t / x A-fragments once (ISA 16-bit A 16x32 layout) --------
    union { v16h v; unsigned u[8]; } ft[2], fx[2];
    #pragma unroll
    for (int mt = 0; mt < 2; ++mt) {
      const int row = mbase + mt * 16 + lm;
      #pragma unroll
      for (int v = 0; v < 8; ++v) {
        const int k = ((v >> 2) << 4) + koffA + ((v & 3) << 1);
        ft[mt].u[v] = *(const unsigned*)&sTh[row][k];
        fx[mt].u[v] = *(const unsigned*)&sXh[row][k];
      }
    }

    // ---- s = 0: residual (raw x) ------------------------------------------
    #pragma unroll
    for (int nt = 0; nt < 4; ++nt) {
      v16h b = ldB(0, nt);
      #pragma unroll
      for (int mt = 0; mt < 2; ++mt)
        acc[mt][nt] = __builtin_amdgcn_wmma_f32_16x16x32_f16(
            false, fx[mt].v, false, b, (short)0, acc[mt][nt], false, false);
    }
    // ---- s = 1: T_1 = t ----------------------------------------------------
    #pragma unroll
    for (int nt = 0; nt < 4; ++nt) {
      v16h b = ldB(1, nt);
      #pragma unroll
      for (int mt = 0; mt < 2; ++mt)
        acc[mt][nt] = __builtin_amdgcn_wmma_f32_16x16x32_f16(
            false, ft[mt].v, false, b, (short)0, acc[mt][nt], false, false);
    }
    // ---- s = 2..10: packed-f16 Chebyshev recurrence in registers ----------
    v16h Tm2[2], Tm1[2], t2[2];
    #pragma unroll
    for (int mt = 0; mt < 2; ++mt) {
      Tm2[mt] = splat16((_Float16)1.0f);
      Tm1[mt] = ft[mt].v;
      t2[mt]  = ft[mt].v + ft[mt].v;
    }
    #pragma unroll
    for (int s = 2; s <= 10; ++s) {
      v16h Tc[2];
      #pragma unroll
      for (int mt = 0; mt < 2; ++mt)
        Tc[mt] = t2[mt] * Tm1[mt] - Tm2[mt];
      #pragma unroll
      for (int nt = 0; nt < 4; ++nt) {
        v16h b = ldB(s, nt);
        #pragma unroll
        for (int mt = 0; mt < 2; ++mt)
          acc[mt][nt] = __builtin_amdgcn_wmma_f32_16x16x32_f16(
              false, Tc[mt], false, b, (short)0, acc[mt][nt], false, false);
      }
      #pragma unroll
      for (int mt = 0; mt < 2; ++mt) {
        Tm2[mt] = Tm1[mt];
        Tm1[mt] = Tc[mt];
      }
    }
  }

  // ---- epilogue: add bias, store (C frag: VGPR r -> M = r + khalf*8) ------
  #pragma unroll
  for (int nt = 0; nt < 4; ++nt) {
    const int gn = n0 + nt * 16 + lm;
    const float bv = bias[gn];
    #pragma unroll
    for (int mt = 0; mt < 2; ++mt) {
      const int gmb = m0 + mbase + mt * 16 + khalf * 8;
      #pragma unroll
      for (int r = 0; r < 8; ++r)
        out[(size_t)(gmb + r) * OUTDIM + gn] = acc[mt][nt][r] + bv;
    }
  }
}

// ---------------------------------------------------------------------------
extern "C" void kernel_launch(void* const* d_in, const int* in_sizes, int n_in,
                              void* d_out, int out_size, void* d_ws, size_t ws_size,
                              hipStream_t stream) {
  const float* x      = (const float*)d_in[0];   // [8192,1024]
  const float* coeffs = (const float*)d_in[1];   // [1024,1024,11]
  const float* basew  = (const float*)d_in[2];   // [1024,1024]
  float* out = (float*)d_out;

  _Float16* wBt = (_Float16*)d_ws;                                   // 22 MB
  float* bias = (float*)((char*)d_ws + (size_t)NSEG * OUTDIM * INDIM * 2);

  dim3 tg(INDIM / 64, OUTDIM / 64);
  cheb_transform_weights<<<tg, 256, 0, stream>>>(coeffs, basew, wBt);
  cheb_bias<<<OUTDIM, 256, 0, stream>>>(coeffs, bias);

  dim3 gg(OUTDIM / BN, BDIM / BM);
  cheb_gemm<<<gg, NTHREADS, 0, stream>>>(x, wBt, bias, out);
}
